// HiPPO_47287589929889
// MI455X (gfx1250) — matrix operand order repaired and checked by reading, not compiled, not run on hardware
//
#include <hip/hip_runtime.h>

// ---------------------------------------------------------------------------
// HiPPO-LegS scan:  c_t = c_{t-1} @ dA^T + f_t * dB,   outputs all c_t.
// L=2048, B=128, N=256, fp32.
//
// Chunked parallel scan (constant transition matrix):
//   pass 1: local scans per chunk (zero init) -> chunk end states z_j
//   pass 2: tiny sequential boundary scan with P = dA^64 (repeated squaring)
//   pass 3: replay each chunk from its true boundary state, write outputs
//
// GEMM core: fp32 state, split-bf16 (hi+lo) operands, 3x
// v_wmma_f32_16x16x32_bf16 per 16x16x32 tile (~fp32 accuracy, 2.7x the
// instruction density of the K=4 fp32 WMMA). Ping-pong LDS state buffer
// (1 barrier/step) + 1-deep A-fragment software pipeline.
// ---------------------------------------------------------------------------

#define LSEQ 2048
#define BB   128
#define NN   256
#define TC   64
#define NCHUNK (LSEQ / TC)   // 32
#define BT   (BB / 16)       // 8 batch tiles of 16 rows

typedef __attribute__((ext_vector_type(16))) __bf16 v16bf;
typedef __attribute__((ext_vector_type(8)))  __bf16 v8bf;
typedef __attribute__((ext_vector_type(8)))  float  v8f;

union V16U { v16bf v; v8bf h[2]; };

// ---- dA^(2^k) by repeated squaring (tiny: 16.8 MFLOP per launch) ----------
__global__ void matsquare_kernel(const float* __restrict__ src,
                                 float* __restrict__ dst) {
  int i = blockIdx.x;   // row
  int j = threadIdx.x;  // col
  float acc = 0.f;
  #pragma unroll 8
  for (int k = 0; k < NN; ++k)
    acc = fmaf(src[i * NN + k], src[k * NN + j], acc);
  dst[i * NN + j] = acc;
}

// ---- split fp32 matrix into bf16 hi + bf16 lo residual --------------------
__global__ void split_bf16_kernel(const float* __restrict__ a,
                                  __bf16* __restrict__ hi,
                                  __bf16* __restrict__ lo) {
  int i = blockIdx.x * blockDim.x + threadIdx.x;
  float v = a[i];
  __bf16 h = (__bf16)v;
  hi[i] = h;
  lo[i] = (__bf16)(v - (float)h);
}

// ---- sequential scan over chunk boundaries (rows independent) -------------
// C_j = C_{j-1} @ P^T + z_j ;  stores the state ENTERING chunk j.
__global__ void boundary_scan_kernel(const float* __restrict__ P,   // dA^TC, [n][k]
                                     const float* __restrict__ z,   // [NCHUNK][B][N]
                                     float* __restrict__ Cb) {      // [NCHUNK][B][N]
  int b = blockIdx.x;      // 0..127
  int n = threadIdx.x;     // 0..255
  __shared__ float row[NN];
  float c = 0.f;
  for (int j = 0; j < NCHUNK; ++j) {
    Cb[((size_t)j * BB + b) * NN + n] = c;
    row[n] = c;
    __syncthreads();
    float acc = z[((size_t)j * BB + b) * NN + n];
    #pragma unroll 8
    for (int k = 0; k < NN; ++k)
      acc = fmaf(row[k], P[n * NN + k], acc);
    __syncthreads();
    c = acc;
  }
}

// ---- main WMMA scan kernel ------------------------------------------------
// grid = NCHUNK * BT blocks, 512 threads (16 waves). Wave w owns output
// column block w (16 cols). State tile (16 rows x 256 cols) lives in fp32
// accumulators, shared between waves each step via bf16 hi/lo LDS buffers.
#define ROWSTRIDE 264   // 16B-aligned, shifts banks by 4 per row -> conflict free

template <bool FINAL>
__global__ __launch_bounds__(512)
void hippo_scan_kernel(const float*  __restrict__ inputs,  // [L][B]
                       const __bf16* __restrict__ Ahi,     // dA hi, [n][k]
                       const __bf16* __restrict__ Alo,     // dA lo, [n][k]
                       const float*  __restrict__ dB,      // [N]
                       const float*  __restrict__ Cb,      // boundary states (FINAL)
                       float*        __restrict__ zb,      // chunk end states (!FINAL)
                       float*        __restrict__ out) {   // [L][B][N]   (FINAL)
  // ping-pong state buffers -> single barrier per time step
  __shared__ __align__(16) __bf16 sh_hi[2][16][ROWSTRIDE];
  __shared__ __align__(16) __bf16 sh_lo[2][16][ROWSTRIDE];
  __shared__ float f_sh[TC][16];

  const int blk  = blockIdx.x;
  const int j    = blk >> 3;          // chunk index
  const int bt   = blk & 7;           // batch tile
  const int wave = threadIdx.x >> 5;  // 0..15 == output column block
  const int lane = threadIdx.x & 31;
  const int lm   = lane & 15;         // A: row | B: col | C/D: col
  const int lh   = lane >> 4;         // half select
  const int cn   = wave * 16 + lm;    // global column of this lane (B & C/D)

  // stage this chunk's inputs (TC x 16 floats)
  for (int idx = threadIdx.x; idx < TC * 16; idx += 512) {
    int tt = idx >> 4, rr = idx & 15;
    f_sh[tt][rr] = inputs[(j * TC + tt) * BB + bt * 16 + rr];
  }

  // preload constant B fragments: dA stored [n][k] -> 16 contiguous K per lane
  v16bf Bhi[8], Blo[8];
  #pragma unroll
  for (int kb = 0; kb < 8; ++kb) {
    const __bf16* ph = Ahi + (size_t)cn * NN + kb * 32 + lh * 16;
    const __bf16* pl = Alo + (size_t)cn * NN + kb * 32 + lh * 16;
    V16U th, tl2;
    th.h[0]  = *(const v8bf*)ph;  th.h[1]  = *(const v8bf*)(ph + 8);
    tl2.h[0] = *(const v8bf*)pl;  tl2.h[1] = *(const v8bf*)(pl + 8);
    Bhi[kb] = th.v;
    Blo[kb] = tl2.v;
  }
  const float dBreg = dB[cn];

  // fp32 state tile accumulator (C/D layout: acc[r] -> row r+8*lh, col cn)
  v8f acc;
  if (FINAL) {
    #pragma unroll
    for (int r = 0; r < 8; ++r)
      acc[r] = Cb[((size_t)j * BB + bt * 16 + r + 8 * lh) * NN + cn];
  } else {
    #pragma unroll
    for (int r = 0; r < 8; ++r) acc[r] = 0.f;
  }

  for (int tl = 0; tl < TC; ++tl) {
    const int p = tl & 1;

    // publish c_{t-1} as bf16 hi/lo into buffer p
    #pragma unroll
    for (int r = 0; r < 8; ++r) {
      float v = acc[r];
      __bf16 hv = (__bf16)v;
      sh_hi[p][r + 8 * lh][cn] = hv;
      sh_lo[p][r + 8 * lh][cn] = (__bf16)(v - (float)hv);
    }
    __syncthreads();   // only barrier this step: reads of buf[p] at step t are
                       // separated from the next write of buf[p] (step t+2) by
                       // the step t+1 barrier.

    // three independent WMMA chains: hi*hi (seeded with input term), hi*lo, lo*hi
    v8f ahh, ahl, alh;
    #pragma unroll
    for (int r = 0; r < 8; ++r) {
      ahh[r] = f_sh[tl][r + 8 * lh] * dBreg;
      ahl[r] = 0.f;
      alh[r] = 0.f;
    }

    // 1-deep software pipeline on A-fragments (ds_load_b128 x4 per stage)
    V16U ah[2], al[2];
    {
      const __bf16* pah = &sh_hi[p][lm][16 * lh];
      const __bf16* pal = &sh_lo[p][lm][16 * lh];
      ah[0].h[0] = *(const v8bf*)pah;  ah[0].h[1] = *(const v8bf*)(pah + 8);
      al[0].h[0] = *(const v8bf*)pal;  al[0].h[1] = *(const v8bf*)(pal + 8);
    }
    #pragma unroll
    for (int kb = 0; kb < 8; ++kb) {
      const int cur = kb & 1;
      const int nxt = cur ^ 1;
      if (kb < 7) {
        const __bf16* pah = &sh_hi[p][lm][(kb + 1) * 32 + 16 * lh];
        const __bf16* pal = &sh_lo[p][lm][(kb + 1) * 32 + 16 * lh];
        ah[nxt].h[0] = *(const v8bf*)pah;  ah[nxt].h[1] = *(const v8bf*)(pah + 8);
        al[nxt].h[0] = *(const v8bf*)pal;  al[nxt].h[1] = *(const v8bf*)(pal + 8);
      }
      ahh = __builtin_amdgcn_wmma_f32_16x16x32_bf16(false, ah[cur].v, false, Bhi[kb],
                                                    (short)0, ahh, false, false);
      ahl = __builtin_amdgcn_wmma_f32_16x16x32_bf16(false, ah[cur].v, false, Blo[kb],
                                                    (short)0, ahl, false, false);
      alh = __builtin_amdgcn_wmma_f32_16x16x32_bf16(false, al[cur].v, false, Bhi[kb],
                                                    (short)0, alh, false, false);
    }

    #pragma unroll
    for (int r = 0; r < 8; ++r) acc[r] = ahh[r] + (ahl[r] + alh[r]);

    if (FINAL) {
      const size_t t = (size_t)j * TC + tl;
      float* op = out + (t * BB + bt * 16) * NN;
      #pragma unroll
      for (int r = 0; r < 8; ++r)
        op[(r + 8 * lh) * NN + cn] = acc[r];
    }
  }

  if (!FINAL) {
    #pragma unroll
    for (int r = 0; r < 8; ++r)
      zb[((size_t)j * BB + bt * 16 + r + 8 * lh) * NN + cn] = acc[r];
  }
}

// ---------------------------------------------------------------------------
extern "C" void kernel_launch(void* const* d_in, const int* in_sizes, int n_in,
                              void* d_out, int out_size, void* d_ws, size_t ws_size,
                              hipStream_t stream) {
  const float* inputs = (const float*)d_in[0];   // (L, B)
  const float* dA     = (const float*)d_in[1];   // (N, N)
  const float* dB     = (const float*)d_in[2];   // (N,)
  float*       out    = (float*)d_out;           // (L, B, N)

  char* ws = (char*)d_ws;
  float*  m0  = (float*)(ws);                         // 256*256*4 = 256 KB
  float*  m1  = (float*)(ws + 262144);                // 256 KB
  __bf16* Ahi = (__bf16*)(ws + 524288);               // 128 KB
  __bf16* Alo = (__bf16*)(ws + 655360);               // 128 KB
  float*  zb  = (float*)(ws + 786432);                // 32*128*256*4 = 4 MB
  float*  Cb  = (float*)(ws + 786432 + 4194304);      // 4 MB

  // P = dA^TC via repeated squaring (TC = 64 = 2^6)
  hipMemcpyAsync(m0, dA, (size_t)NN * NN * sizeof(float),
                 hipMemcpyDeviceToDevice, stream);
  float* src = m0;
  float* dst = m1;
  for (int s = 0; s < 6; ++s) {
    matsquare_kernel<<<NN, NN, 0, stream>>>(src, dst);
    float* t = src; src = dst; dst = t;
  }
  // result (dA^64) now in `src`

  split_bf16_kernel<<<(NN * NN) / 256, 256, 0, stream>>>(dA, Ahi, Alo);

  // pass 1: local chunk scans -> z_j
  hippo_scan_kernel<false><<<NCHUNK * BT, 512, 0, stream>>>(
      inputs, Ahi, Alo, dB, nullptr, zb, nullptr);

  // pass 2: sequential boundary scan -> Cb (state entering each chunk)
  boundary_scan_kernel<<<BB, NN, 0, stream>>>(src, zb, Cb);

  // pass 3: replay with true boundary states, write all outputs
  hippo_scan_kernel<true><<<NCHUNK * BT, 512, 0, stream>>>(
      inputs, Ahi, Alo, dB, Cb, nullptr, out);
}
